// GoEModel_22565758173211
// MI455X (gfx1250) — compile-verified, matrix operand order checked by reading, not run on hardware
//
#include <hip/hip_runtime.h>
#include <hip/hip_bf16.h>
#include <math.h>

// ---------------- model dims ----------------
#define V_  32000
#define D_  512
#define E_  4
#define H_  8
#define DH_ 64
#define FF_ 2048
#define L_  2
#define RH_ 256
#define B_  8
#define S_  128
#define MAXV_ 2
#define SQRTD_ 22.62741699796952f

typedef __attribute__((ext_vector_type(16))) __bf16 bf16x16;
typedef __attribute__((ext_vector_type(8)))  __bf16 bf16x8;
typedef __attribute__((ext_vector_type(4)))  __bf16 bf16x4;
typedef __attribute__((ext_vector_type(8)))  float  v8f;
typedef __attribute__((ext_vector_type(4)))  unsigned int u32x4;
typedef __attribute__((ext_vector_type(4)))  int i32x4;
typedef __attribute__((ext_vector_type(8)))  int i32x8;

#if defined(__has_builtin)
#if __has_builtin(__builtin_amdgcn_tensor_load_to_lds) && __has_builtin(__builtin_amdgcn_s_wait_tensorcnt)
#define USE_TDM 1
#endif
#endif
#ifndef USE_TDM
#define USE_TDM 0
#endif

// ---------------- helpers ----------------
__device__ __forceinline__ float sigmoidf_(float x) { return 1.0f / (1.0f + expf(-x)); }
__device__ __forceinline__ float geluf_(float x)    { return 0.5f * x * (1.0f + erff(x * 0.70710678118654752f)); }

// Load a 16x32 (rows x K) bf16 fragment from an LDS tile stored row-major ld=32.
// Layout per CDNA5 ISA 7.12.2 (16-bit A matrix): lane = half*16 + li, row = li,
// vgpr r holds k pair {((r&3)*2 + (r>>2)*16 + half*8), +1}.
__device__ __forceinline__ bf16x16 load_frag(const __bf16* base, int row, int half) {
  bf16x16 f;
#pragma unroll
  for (int r = 0; r < 8; ++r) {
    int k = ((r & 3) << 1) + ((r >> 2) << 4) + (half << 3);
    f[2 * r]     = base[row * 32 + k];
    f[2 * r + 1] = base[row * 32 + k + 1];
  }
  return f;
}

// ---------------- fp32 -> bf16 weight conversion (vector) ----------------
__global__ __launch_bounds__(256) void cvt_kernel(const float* __restrict__ in, __bf16* __restrict__ out) {
  size_t i = ((size_t)blockIdx.x * 256 + threadIdx.x) * 4;
  float4 f = *(const float4*)&in[i];
  bf16x4 o;
  o[0] = (__bf16)f.x; o[1] = (__bf16)f.y; o[2] = (__bf16)f.z; o[3] = (__bf16)f.w;
  *(bf16x4*)&out[i] = o;
}

// ---------------- generic bf16-WMMA GEMM:  C = epi(A @ W^T + bias) ----------------
// A: fp32 [M,K] per-sample (stride sA).  Wt: bf16 [N,K] per-expert (stride sW, elements).
// EPI: 0=none 1=sigmoid 2=gelu 3=add into existing C (residual)
#define TM 64
#define TN 64
#define TK 32
template <int EPI>
__global__ __launch_bounds__(128) void gemm_kernel(
    const float* __restrict__ A, const __bf16* __restrict__ Wt,
    const float* __restrict__ bias, float* __restrict__ C,
    int M, int N, int K, long sA, long sW, long sB, long sC,
    const int* __restrict__ eidx, const int* __restrict__ mask) {
  int z = blockIdx.z;
  if (mask && !mask[z]) return;
  int e = eidx ? eidx[z] : 0;
  A  += (long)z * sA;
  Wt += (long)e * sW;
  const float* bp = bias ? bias + (long)e * sB : nullptr;
  C += (long)z * sC;

  __shared__ __bf16 As[TM * TK];
  __shared__ __bf16 Ws[TN * TK];

  int tid  = threadIdx.x;
  int lane = tid & 31, wave = tid >> 5;
  int half = lane >> 4, li = lane & 15;
  int m0 = blockIdx.y * TM;
  int n0 = blockIdx.x * TN;
  int wm = (wave >> 1) * 32, wn = (wave & 1) * 32;

  v8f acc[2][2] = {};

  for (int k0 = 0; k0 < K; k0 += TK) {
    // ---- stage A tile (fp32 -> bf16), 512 float4 chunks, 4 per thread; batch loads ----
    float4 fa[4];
#pragma unroll
    for (int i = 0; i < 4; ++i) {
      int v = tid + i * 128;
      int r = v >> 3, c = (v & 7) * 4;
      fa[i] = *(const float4*)&A[(size_t)(m0 + r) * K + (k0 + c)];
    }
#pragma unroll
    for (int i = 0; i < 4; ++i) {
      int v = tid + i * 128;
      int r = v >> 3, c = (v & 7) * 4;
      bf16x4 o;
      o[0] = (__bf16)fa[i].x; o[1] = (__bf16)fa[i].y; o[2] = (__bf16)fa[i].z; o[3] = (__bf16)fa[i].w;
      *(bf16x4*)&As[r * TK + c] = o;
    }
    if (k0 + TK < K)
      __builtin_prefetch(&A[(size_t)(m0 + (tid & 63)) * K + k0 + TK], 0, 1);

#if USE_TDM
    // ---- stage W tile via Tensor Data Mover: 64 rows x 32 cols of bf16 ----
    if (wave == 0) {
      unsigned lds_off = (unsigned)(uintptr_t)(void*)Ws;  // shared aperture: low 32 bits == LDS byte addr
      unsigned long long ga = (unsigned long long)(uintptr_t)(const void*)&Wt[(size_t)n0 * K + k0];
      u32x4 g0;
      g0[0] = 1u;                                           // count=1, user mode
      g0[1] = lds_off;                                      // lds_addr
      g0[2] = (unsigned)ga;                                 // global_addr[31:0]
      g0[3] = (unsigned)((ga >> 32) & 0x1FFFFFFull) | (2u << 30);  // global_addr[56:32] | type=2
      i32x8 g1;
      g1[0] = 0x10000;                // workgroup_mask=0, data_size=1 (2B)
      g1[1] = 0;                      // no atomic barrier; tensor_dim0[15:0]=0 (dim0 = 1<<30)
      g1[2] = 0x4000;                 // tensor_dim0[31:16]=0x4000; tensor_dim1[15:0]=0
      g1[3] = 0x10 | (TK << 16);      // tensor_dim1[31:16]=0x10 (dim1 = 1<<20); tile_dim0=32
      g1[4] = TN;                     // tile_dim1=64 rows; tile_dim2=0
      g1[5] = K;                      // tensor_dim0_stride[31:0] (elements)
      g1[6] = 0;                      // stride[47:32]=0; tensor_dim1_stride[15:0]=0
      g1[7] = 0;
      i32x4 gz = {0, 0, 0, 0};
#if __clang_major__ >= 23
      i32x8 gz8 = {0, 0, 0, 0, 0, 0, 0, 0};
      __builtin_amdgcn_tensor_load_to_lds(g0, g1, gz, gz, gz8, 0);
#else
      __builtin_amdgcn_tensor_load_to_lds(g0, g1, gz, gz, 0);
#endif
      __builtin_amdgcn_s_wait_tensorcnt(0);
    }
#else
    // ---- manual vectorized W staging (bf16, 16B chunks, 2 per thread) ----
    bf16x8 wb[2];
#pragma unroll
    for (int i = 0; i < 2; ++i) {
      int v = tid + i * 128;
      int r = v >> 2, c = (v & 3) * 8;
      wb[i] = *(const bf16x8*)&Wt[(size_t)(n0 + r) * K + (k0 + c)];
    }
#pragma unroll
    for (int i = 0; i < 2; ++i) {
      int v = tid + i * 128;
      int r = v >> 2, c = (v & 3) * 8;
      *(bf16x8*)&Ws[r * TK + c] = wb[i];
    }
#endif
    __syncthreads();

    bf16x16 a0 = load_frag(As + (wm +  0) * 32, li, half);
    bf16x16 a1 = load_frag(As + (wm + 16) * 32, li, half);
    bf16x16 b0 = load_frag(Ws + (wn +  0) * 32, li, half);
    bf16x16 b1 = load_frag(Ws + (wn + 16) * 32, li, half);

    acc[0][0] = __builtin_amdgcn_wmma_f32_16x16x32_bf16(false, a0, false, b0, (short)0, acc[0][0], false, false);
    acc[0][1] = __builtin_amdgcn_wmma_f32_16x16x32_bf16(false, a0, false, b1, (short)0, acc[0][1], false, false);
    acc[1][0] = __builtin_amdgcn_wmma_f32_16x16x32_bf16(false, a1, false, b0, (short)0, acc[1][0], false, false);
    acc[1][1] = __builtin_amdgcn_wmma_f32_16x16x32_bf16(false, a1, false, b1, (short)0, acc[1][1], false, false);
    __syncthreads();
  }

  // C/D layout: lanes 0-15 -> M=r, lanes 16-31 -> M=r+8; N = li
#pragma unroll
  for (int i = 0; i < 2; ++i)
#pragma unroll
    for (int j = 0; j < 2; ++j) {
      int n = n0 + wn + 16 * j + li;
      float bv = bp ? bp[n] : 0.0f;
#pragma unroll
      for (int r = 0; r < 8; ++r) {
        int m = m0 + wm + 16 * i + r + half * 8;
        float v = acc[i][j][r] + bv;
        if (EPI == 1) v = sigmoidf_(v);
        else if (EPI == 2) v = geluf_(v);
        size_t o = (size_t)m * N + n;
        if (EPI == 3) v += C[o];
        C[o] = v;
      }
    }
}

// ---------------- embed + pos-encode + input LN (writes x and final_repr) ----------------
__global__ __launch_bounds__(256) void embed_kernel(
    const int* __restrict__ ids, const float* __restrict__ emb, const float* __restrict__ pe,
    const float* __restrict__ g, const float* __restrict__ be,
    float* __restrict__ x, float* __restrict__ frepr) {
  int row = blockIdx.x;         // over B*S
  int s = row & (S_ - 1);
  int tid = threadIdx.x;
  __shared__ float red[256];
  int id = ids[row];
  float v0 = emb[(size_t)id * D_ + tid]       * SQRTD_ + pe[(size_t)s * D_ + tid];
  float v1 = emb[(size_t)id * D_ + tid + 256] * SQRTD_ + pe[(size_t)s * D_ + tid + 256];
  red[tid] = v0 + v1; __syncthreads();
  for (int t = 128; t > 0; t >>= 1) { if (tid < t) red[tid] += red[tid + t]; __syncthreads(); }
  float m = red[0] / D_; __syncthreads();
  float d0 = v0 - m, d1 = v1 - m;
  red[tid] = d0 * d0 + d1 * d1; __syncthreads();
  for (int t = 128; t > 0; t >>= 1) { if (tid < t) red[tid] += red[tid + t]; __syncthreads(); }
  float inv = rsqrtf(red[0] / D_ + 1e-5f);
  size_t off = (size_t)row * D_;
  float o0 = d0 * inv * g[tid] + be[tid];
  float o1 = d1 * inv * g[tid + 256] + be[tid + 256];
  x[off + tid] = o0;       x[off + tid + 256] = o1;
  frepr[off + tid] = o0;   frepr[off + tid + 256] = o1;
}

// ---------------- generic row LayerNorm ----------------
__global__ __launch_bounds__(256) void ln_kernel(
    const float* __restrict__ in, float* __restrict__ out,
    const float* __restrict__ g0, const float* __restrict__ b0, long gs,
    const int* __restrict__ eidx, const int* __restrict__ mask) {
  int row = blockIdx.x;
  int b = row >> 7;             // row / S
  if (mask && !mask[b]) return;
  int e = eidx ? eidx[b] : 0;
  const float* g  = g0 + (size_t)e * gs;
  const float* bb = b0 + (size_t)e * gs;
  int tid = threadIdx.x;
  __shared__ float red[256];
  size_t off = (size_t)row * D_;
  float v0 = in[off + tid], v1 = in[off + tid + 256];
  red[tid] = v0 + v1; __syncthreads();
  for (int t = 128; t > 0; t >>= 1) { if (tid < t) red[tid] += red[tid + t]; __syncthreads(); }
  float m = red[0] / D_; __syncthreads();
  float d0 = v0 - m, d1 = v1 - m;
  red[tid] = d0 * d0 + d1 * d1; __syncthreads();
  for (int t = 128; t > 0; t >>= 1) { if (tid < t) red[tid] += red[tid + t]; __syncthreads(); }
  float inv = rsqrtf(red[0] / D_ + 1e-5f);
  out[off + tid]       = d0 * inv * g[tid]       + bb[tid];
  out[off + tid + 256] = d1 * inv * g[tid + 256] + bb[tid + 256];
}

// ---------------- attention (per b,h,q-row): scores, softmax, context ----------------
__global__ __launch_bounds__(128) void attn_kernel(
    const float* __restrict__ qkv, float* __restrict__ attno, const int* __restrict__ mask) {
  int b = blockIdx.z; if (!mask[b]) return;
  int h = blockIdx.y, q = blockIdx.x;
  int tid = threadIdx.x;                    // one thread per key (S=128)
  __shared__ float sc[S_];
  __shared__ float red[128];
  const float* base = qkv + (size_t)b * S_ * 3 * D_;
  const float* qv = base + (size_t)q   * 3 * D_ + h * DH_;
  const float* kv = base + (size_t)tid * 3 * D_ + D_ + h * DH_;
  float dot = 0.0f;
#pragma unroll 8
  for (int d = 0; d < DH_; ++d) dot += qv[d] * kv[d];
  dot *= 0.125f;                            // 1/sqrt(64)
  red[tid] = dot; __syncthreads();
  for (int t = 64; t > 0; t >>= 1) { if (tid < t) red[tid] = fmaxf(red[tid], red[tid + t]); __syncthreads(); }
  float mx = red[0]; __syncthreads();
  float ex = expf(dot - mx);
  red[tid] = ex; __syncthreads();
  for (int t = 64; t > 0; t >>= 1) { if (tid < t) red[tid] += red[tid + t]; __syncthreads(); }
  float inv = 1.0f / red[0]; __syncthreads();
  sc[tid] = ex * inv; __syncthreads();
  if (tid < DH_) {
    const float* vv = base + 2 * D_ + h * DH_ + tid;
    float acc = 0.0f;
    for (int k = 0; k < S_; ++k) acc += sc[k] * vv[(size_t)k * 3 * D_];
    attno[((size_t)b * S_ + q) * D_ + h * DH_ + tid] = acc;
  }
}

// ---------------- gated-attention residual: x += ln(h + g*a) ----------------
__global__ __launch_bounds__(256) void ga_kernel(
    const float* __restrict__ hb, const float* __restrict__ gb, const float* __restrict__ ab,
    float* __restrict__ x, const float* __restrict__ g0, const float* __restrict__ b0, long gs,
    const int* __restrict__ eidx, const int* __restrict__ mask) {
  int row = blockIdx.x; int b = row >> 7;
  if (!mask[b]) return;
  int e = eidx[b];
  const float* g  = g0 + (size_t)e * gs;
  const float* bb = b0 + (size_t)e * gs;
  int tid = threadIdx.x;
  __shared__ float red[256];
  size_t off = (size_t)row * D_;
  float v0 = hb[off + tid]       + gb[off + tid]       * ab[off + tid];
  float v1 = hb[off + tid + 256] + gb[off + tid + 256] * ab[off + tid + 256];
  red[tid] = v0 + v1; __syncthreads();
  for (int t = 128; t > 0; t >>= 1) { if (tid < t) red[tid] += red[tid + t]; __syncthreads(); }
  float m = red[0] / D_; __syncthreads();
  float d0 = v0 - m, d1 = v1 - m;
  red[tid] = d0 * d0 + d1 * d1; __syncthreads();
  for (int t = 128; t > 0; t >>= 1) { if (tid < t) red[tid] += red[tid + t]; __syncthreads(); }
  float inv = rsqrtf(red[0] / D_ + 1e-5f);
  x[off + tid]       += d0 * inv * g[tid]       + bb[tid];
  x[off + tid + 256] += d1 * inv * g[tid + 256] + bb[tid + 256];
}

// ---------------- concat [h, a] ----------------
__global__ __launch_bounds__(256) void concat_kernel(
    const float* __restrict__ hb, const float* __restrict__ ab, float* __restrict__ cat,
    const int* __restrict__ mask) {
  size_t i = (size_t)blockIdx.x * 256 + threadIdx.x;   // over B*S*2D
  int b = (int)(i / ((size_t)S_ * 2 * D_));
  if (!mask[b]) return;
  size_t row = i / (2 * D_);
  int d = (int)(i % (2 * D_));
  cat[i] = (d < D_) ? hb[row * D_ + d] : ab[row * D_ + d - D_];
}

// ---------------- expert global gate: gv = sigmoid(mean_s(x) . gl_w + gl_b) ----------------
__global__ __launch_bounds__(256) void glstat_kernel(
    const float* __restrict__ x, const float* __restrict__ glw, const float* __restrict__ glb,
    float* __restrict__ gl, const int* __restrict__ eidx, const int* __restrict__ mask) {
  int b = blockIdx.x; if (!mask[b]) return;
  int e = eidx[b]; int tid = threadIdx.x;
  __shared__ float red[256];
  float part = 0.0f;
#pragma unroll
  for (int rep = 0; rep < 2; ++rep) {
    int d = tid + rep * 256;
    float sum = 0.0f;
    for (int s = 0; s < S_; ++s) sum += x[((size_t)b * S_ + s) * D_ + d];
    part += (sum / S_) * glw[(size_t)e * D_ + d];
  }
  red[tid] = part; __syncthreads();
  for (int t = 128; t > 0; t >>= 1) { if (tid < t) red[tid] += red[tid + t]; __syncthreads(); }
  if (tid == 0) gl[b] = sigmoidf_(red[0] + glb[e]);
}

__global__ __launch_bounds__(256) void glmix_kernel(
    const float* __restrict__ xin, const float* __restrict__ tag, const float* __restrict__ gl,
    float* __restrict__ x, const int* __restrict__ eidx, const int* __restrict__ mask) {
  size_t i = (size_t)blockIdx.x * 256 + threadIdx.x;   // over B*S*D
  int b = (int)(i / ((size_t)S_ * D_));
  if (!mask[b]) return;
  int d = (int)(i % D_);
  float gv = gl[b];
  x[i] = gv * x[i] + (1.0f - gv) * xin[i] + tag[(size_t)eidx[b] * D_ + d];
}

// ---------------- copies ----------------
__global__ __launch_bounds__(256) void copy_kernel(const float* __restrict__ s, float* __restrict__ d) {
  size_t i = (size_t)blockIdx.x * 256 + threadIdx.x; d[i] = s[i];
}
__global__ __launch_bounds__(256) void mcopy_kernel(const float* __restrict__ s, float* __restrict__ d,
                                                    const int* __restrict__ flag) {
  size_t i = (size_t)blockIdx.x * 256 + threadIdx.x;
  int b = (int)(i / ((size_t)S_ * D_));
  if (!flag[b]) return;
  d[i] = s[i];
}

// ---------------- router state init ----------------
__global__ void init_kernel(int* st) {
  int i = threadIdx.x;
  if (i < B_) { st[i] = 0; st[B_ + i] = 0; st[2 * B_ + i] = 0; st[3 * B_ + i] = 1; } // eidx,smask,tfin,active
  if (i < B_ * E_) st[4 * B_ + i] = 0;                                               // visits
}

// ---------------- router (one block per sample) ----------------
__global__ __launch_bounds__(256) void router_kernel(
    const float* __restrict__ x,
    const float* __restrict__ fc1w, const float* __restrict__ fc1b,
    const float* __restrict__ fc2w, const float* __restrict__ fc2b,
    const float* __restrict__ ng, const float* __restrict__ nb, const float* __restrict__ qq,
    int* __restrict__ st) {
  int b = blockIdx.x; int tid = threadIdx.x;
  int* eidx = st; int* smask = st + B_; int* tfin = st + 2 * B_;
  int* act = st + 3 * B_; int* visits = st + 4 * B_;
  __shared__ float summ[2 * D_];
  __shared__ float hR[RH_];
  __shared__ float lg[8];
#pragma unroll
  for (int rep = 0; rep < 2; ++rep) {
    int d = tid + rep * 256;
    float sum = 0.0f, mx = -3.0e38f;
    for (int s = 0; s < S_; ++s) {
      float v = x[((size_t)b * S_ + s) * D_ + d];
      sum += v; mx = fmaxf(mx, v);
    }
    summ[d] = sum / S_;
    summ[D_ + d] = mx;
  }
  __syncthreads();
  {
    float acc = fc1b[tid];
    const float* w = fc1w + (size_t)tid * 2 * D_;
    for (int j = 0; j < 2 * D_; ++j) acc += w[j] * summ[j];
    hR[tid] = geluf_(acc);
  }
  __syncthreads();
  if (tid < E_ + 1) {
    float acc = fc2b[tid] + qq[tid];
    const float* w = fc2w + (size_t)tid * RH_;
    for (int j = 0; j < RH_; ++j) acc += w[j] * hR[j];
    lg[tid] = acc;
  }
  __syncthreads();
  if (tid == 0) {
    float m = 0.0f;
    for (int j = 0; j <= E_; ++j) m += lg[j];
    m /= (E_ + 1);
    float v = 0.0f;
    for (int j = 0; j <= E_; ++j) { float dd = lg[j] - m; v += dd * dd; }
    v /= (E_ + 1);
    float inv = rsqrtf(v + 1e-5f);
    float best = -3.4e38f; int choice = 0;
    for (int j = 0; j <= E_; ++j) {
      float l = (lg[j] - m) * inv * ng[j] + nb[j];
      l = fminf(10.0f, fmaxf(-10.0f, l));
      if (j < E_ && visits[b * E_ + j] >= MAXV_) l = -1e9f;
      if (l > best) { best = l; choice = j; }
    }
    int term = (choice == E_);
    int ei = term ? (E_ - 1) : choice;
    int a = act[b];
    int sm = (a && !term) ? 1 : 0;
    eidx[b] = ei; smask[b] = sm; tfin[b] = (a && term) ? 1 : 0;
    if (sm) visits[b * E_ + ei] += 1;
    act[b] = sm;
  }
}

// ---------------- host orchestration ----------------
extern "C" void kernel_launch(void* const* d_in, const int* in_sizes, int n_in,
                              void* d_out, int out_size, void* d_ws, size_t ws_size,
                              hipStream_t stream) {
  (void)in_sizes; (void)n_in; (void)out_size; (void)ws_size;
  const int*   ids   = (const int*)  d_in[0];
  const float* emb   = (const float*)d_in[1];
  const float* pe    = (const float*)d_in[2];
  const float* ing   = (const float*)d_in[3];
  const float* inb   = (const float*)d_in[4];
  const float* wqkv  = (const float*)d_in[5];
  const float* bqkv  = (const float*)d_in[6];
  const float* wo    = (const float*)d_in[7];
  const float* bo    = (const float*)d_in[8];
  const float* gw    = (const float*)d_in[9];
  const float* gbb   = (const float*)d_in[10];
  const float* gang  = (const float*)d_in[11];
  const float* ganb  = (const float*)d_in[12];
  const float* n1g   = (const float*)d_in[13];
  const float* n1b   = (const float*)d_in[14];
  const float* n2g   = (const float*)d_in[15];
  const float* n2b   = (const float*)d_in[16];
  const float* fw1   = (const float*)d_in[17];
  const float* fb1   = (const float*)d_in[18];
  const float* fw2   = (const float*)d_in[19];
  const float* fb2   = (const float*)d_in[20];
  const float* tag   = (const float*)d_in[21];
  const float* glw   = (const float*)d_in[22];
  const float* glb   = (const float*)d_in[23];
  const float* fc1w  = (const float*)d_in[24];
  const float* fc1b  = (const float*)d_in[25];
  const float* fc2w  = (const float*)d_in[26];
  const float* fc2b  = (const float*)d_in[27];
  const float* rng   = (const float*)d_in[28];
  const float* rnb   = (const float*)d_in[29];
  const float* rq    = (const float*)d_in[30];
  const float* fng   = (const float*)d_in[31];
  const float* fnb   = (const float*)d_in[32];
  const float* lmw   = (const float*)d_in[33];
  const float* lmb   = (const float*)d_in[34];
  float* out = (float*)d_out;

  // ---------------- workspace layout ----------------
  const size_t NX = (size_t)B_ * S_ * D_;           // 524288 floats
  float* W  = (float*)d_ws;
  float* X    = W;                                  // x
  float* XIN  = W + 1 * NX;                         // expert input snapshot
  float* HB   = W + 2 * NX;                         // ln output / scratch
  float* AB   = W + 3 * NX;                         // attention proj output
  float* GB   = W + 4 * NX;                         // gate output
  float* FR   = W + 5 * NX;                         // final_repr
  float* QKV  = W + 6 * NX;                         // [B,S,3D]  (3*NX)
  float* CAT  = W + 9 * NX;                         // [B,S,2D]  (2*NX)
  float* FFH  = W + 11 * NX;                        // [B,S,FF]  (4*NX)
  float* ATT  = W + 15 * NX;                        // attention context [B,S,D]
  float* GL   = W + 16 * NX;                        // [B]
  int*   ST   = (int*)(W + 16 * NX + 16);           // router state ints
  int* eidx = ST; int* smask = ST + B_; int* tfin = ST + 2 * B_; int* act = ST + 3 * B_;

  // bf16 weight copies (converted once per launch)
  const size_t nWqkv = (size_t)E_ * L_ * 3 * D_ * D_;   // 6,291,456
  const size_t nWo   = (size_t)E_ * L_ * D_ * D_;       // 2,097,152
  const size_t nGw   = (size_t)E_ * L_ * D_ * 2 * D_;   // 4,194,304
  const size_t nF1   = (size_t)E_ * L_ * FF_ * D_;      // 8,388,608
  const size_t nF2   = (size_t)E_ * L_ * D_ * FF_;      // 8,388,608
  const size_t nLm   = (size_t)V_ * D_;                 // 16,384,000
  __bf16* wqkvB = (__bf16*)(W + 16 * NX + 4096);
  __bf16* woB   = wqkvB + nWqkv;
  __bf16* gwB   = woB + nWo;
  __bf16* fw1B  = gwB + nGw;
  __bf16* fw2B  = fw1B + nF1;
  __bf16* lmB   = fw2B + nF2;

  const int rowsBlk = B_ * S_;                      // 1024 token rows
  const int elemBlk = (int)(NX / 256);              // 2048 elementwise blocks

  // weight precision conversion (L2-resident afterwards)
  cvt_kernel<<<(int)(nWqkv / 1024), 256, 0, stream>>>(wqkv, wqkvB);
  cvt_kernel<<<(int)(nWo   / 1024), 256, 0, stream>>>(wo,   woB);
  cvt_kernel<<<(int)(nGw   / 1024), 256, 0, stream>>>(gw,   gwB);
  cvt_kernel<<<(int)(nF1   / 1024), 256, 0, stream>>>(fw1,  fw1B);
  cvt_kernel<<<(int)(nF2   / 1024), 256, 0, stream>>>(fw2,  fw2B);
  cvt_kernel<<<(int)(nLm   / 1024), 256, 0, stream>>>(lmw,  lmB);

  init_kernel<<<1, 64, 0, stream>>>(ST);
  embed_kernel<<<rowsBlk, 256, 0, stream>>>(ids, emb, pe, ing, inb, X, FR);

  for (int step = 0; step < 3; ++step) {
    router_kernel<<<B_, 256, 0, stream>>>(X, fc1w, fc1b, fc2w, fc2b, rng, rnb, rq, ST);
    copy_kernel<<<elemBlk, 256, 0, stream>>>(X, XIN);

    for (int l = 0; l < L_; ++l) {
      // h = ln(x, n1)
      ln_kernel<<<rowsBlk, 256, 0, stream>>>(X, HB, n1g + l * D_, n1b + l * D_, (long)L_ * D_, eidx, smask);
      // qkv = h @ wqkv^T + bqkv
      gemm_kernel<0><<<dim3(3 * D_ / TN, S_ / TM, B_), 128, 0, stream>>>(
          HB, wqkvB + (size_t)l * 3 * D_ * D_, bqkv + (size_t)l * 3 * D_, QKV,
          S_, 3 * D_, D_, (long)S_ * D_, (long)L_ * 3 * D_ * D_, (long)L_ * 3 * D_, (long)S_ * 3 * D_, eidx, smask);
      // attention
      attn_kernel<<<dim3(S_, H_, B_), 128, 0, stream>>>(QKV, ATT, smask);
      // a = ctx @ wo^T + bo
      gemm_kernel<0><<<dim3(D_ / TN, S_ / TM, B_), 128, 0, stream>>>(
          ATT, woB + (size_t)l * D_ * D_, bo + (size_t)l * D_, AB,
          S_, D_, D_, (long)S_ * D_, (long)L_ * D_ * D_, (long)L_ * D_, (long)S_ * D_, eidx, smask);
      // g = sigmoid([h,a] @ gate_w^T + gate_b)
      concat_kernel<<<2 * elemBlk, 256, 0, stream>>>(HB, AB, CAT, smask);
      gemm_kernel<1><<<dim3(D_ / TN, S_ / TM, B_), 128, 0, stream>>>(
          CAT, gwB + (size_t)l * D_ * 2 * D_, gbb + (size_t)l * D_, GB,
          S_, D_, 2 * D_, (long)S_ * 2 * D_, (long)L_ * D_ * 2 * D_, (long)L_ * D_, (long)S_ * D_, eidx, smask);
      // x += ln(h + g*a, ga_n)
      ga_kernel<<<rowsBlk, 256, 0, stream>>>(HB, GB, AB, X, gang + l * D_, ganb + l * D_, (long)L_ * D_, eidx, smask);
      // h2 = ln(x, n2); f = gelu(h2@w1^T+b1)@w2^T+b2; x += f
      ln_kernel<<<rowsBlk, 256, 0, stream>>>(X, HB, n2g + l * D_, n2b + l * D_, (long)L_ * D_, eidx, smask);
      gemm_kernel<2><<<dim3(FF_ / TN, S_ / TM, B_), 128, 0, stream>>>(
          HB, fw1B + (size_t)l * FF_ * D_, fb1 + (size_t)l * FF_, FFH,
          S_, FF_, D_, (long)S_ * D_, (long)L_ * FF_ * D_, (long)L_ * FF_, (long)S_ * FF_, eidx, smask);
      gemm_kernel<3><<<dim3(D_ / TN, S_ / TM, B_), 128, 0, stream>>>(
          FFH, fw2B + (size_t)l * D_ * FF_, fb2 + (size_t)l * D_, X,
          S_, D_, FF_, (long)S_ * FF_, (long)L_ * D_ * FF_, (long)L_ * D_, (long)S_ * D_, eidx, smask);
    }
    // expert global gate mix + tag
    glstat_kernel<<<B_, 256, 0, stream>>>(X, glw, glb, GL, eidx, smask);
    glmix_kernel<<<elemBlk, 256, 0, stream>>>(XIN, tag, GL, X, eidx, smask);
    // final_repr for samples that terminated this step
    mcopy_kernel<<<elemBlk, 256, 0, stream>>>(X, FR, tfin);
  }
  // still-active samples
  mcopy_kernel<<<elemBlk, 256, 0, stream>>>(X, FR, act);

  // out = ln(final_repr) @ lm_w^T + lm_b
  ln_kernel<<<rowsBlk, 256, 0, stream>>>(FR, HB, fng, fnb, 0, nullptr, nullptr);
  gemm_kernel<0><<<dim3(V_ / TN, (B_ * S_) / TM, 1), 128, 0, stream>>>(
      HB, lmB, lmb, out, B_ * S_, V_, D_, 0, 0, 0, 0, nullptr, nullptr);
}
